// CRFTagger_9534827397626
// MI455X (gfx1250) — compile-verified, matrix operand order checked by reading, not run on hardware
//
#include <hip/hip_runtime.h>
#include <hip/hip_bf16.h>

typedef __attribute__((ext_vector_type(16))) _Float16 v16h;
typedef __attribute__((ext_vector_type(8)))  _Float16 v8h;
typedef __attribute__((ext_vector_type(8)))  float    v8f;

#define CDIM 128    // classes incl START(126)/STOP(127)
#define TDIM 512
#define BDIM 128
#define ROWS 16     // batch rows per workgroup (= WMMA M)
#define EPAD 144    // f16 row stride (288B = 18*16B, keeps b128 alignment)
#define LOG2E 1.4426950408889634f
#define LN2   0.6931471805599453f

static __device__ __forceinline__ v16h cat8(v8h lo, v8h hi) {
  v16h r;
#pragma unroll
  for (int i = 0; i < 8; ++i) { r[i] = lo[i]; r[i + 8] = hi[i]; }
  return r;
}

// Branchless, NaN-safe select on float bits (cannot be turned into a branch)
static __device__ __forceinline__ float bitsel(int cond, float a, float b) {
  unsigned m = cond ? 0xFFFFFFFFu : 0u;
  return __uint_as_float((__float_as_uint(a) & m) | (__float_as_uint(b) & ~m));
}

// Forward algorithm via exp-rescaled GEMM scan.
// alpha_new_j = feat_j + m + log( sum_i exp(alpha_i - m) * exp(trans_ij) )
__global__ __launch_bounds__(256) void crf_forward_scan(
    const float* __restrict__ feats, const int* __restrict__ maskp,
    const float* __restrict__ trans, float* __restrict__ acc_out)
{
  __shared__ __align__(16) _Float16 sE[CDIM * EPAD];   // E^T[n][k] = exp(trans[k][n])
  __shared__ __align__(16) _Float16 sP[ROWS * EPAD];   // P[m][k]  = exp(alpha - m)
  __shared__ __align__(32) float sAlpha[ROWS * CDIM];  // row-major mirror of alpha
  __shared__ __align__(32) float sMrow[ROWS];

  const int tid   = threadIdx.x;
  const int lane  = tid & 31;
  const int wave  = tid >> 5;        // 0..7 -> N tile index
  const int khalf = lane >> 4;       // 0/1 : K-group selector (ISA A/B layout)
  const int nloc  = lane & 15;       // M row (A) / N col-in-tile (B)
  const int ncol  = wave * 16 + nloc;
  const int rrow  = tid >> 4;        // reduction row: wave w owns rows 2w, 2w+1
  const int rj    = tid & 15;        // 16 lanes per row; owns 8 contiguous columns

  // Precompute E^T = exp(trans)^T in f16 (constant over time)
  for (int idx = tid; idx < CDIM * CDIM; idx += 256) {
    int k = idx >> 7, n = idx & 127;                    // coalesced read over n
    sE[n * EPAD + k] = (_Float16)__builtin_amdgcn_exp2f(trans[k * CDIM + n] * LOG2E);
  }
  // alpha init: each lane owns the 8 elements (M = 8*khalf + v, ncol)
  float alphaReg[8];
#pragma unroll
  for (int v = 0; v < 8; ++v) {
    int M = v + 8 * khalf;
    float a = (ncol == CDIM - 2) ? 0.0f : -10000.0f;
    alphaReg[v] = a;
    sAlpha[M * CDIM + ncol] = a;
  }
  __syncthreads();

  // Preload time-invariant B fragments (column ncol of E, all K) into registers
  v16h bfrag[4];
  {
    const _Float16* er = &sE[ncol * EPAD];
#pragma unroll
    for (int kb = 0; kb < 4; ++kb) {
      v8h lo = *(const v8h*)(er + kb * 32 + 8 * khalf);        // K = 32kb + 8*khalf ..+7
      v8h hi = *(const v8h*)(er + kb * 32 + 16 + 8 * khalf);   // K = +16
      bfrag[kb] = cat8(lo, hi);
    }
  }

  const size_t bbase = (size_t)blockIdx.x * ROWS;

  for (int t = 0; t < TDIM; ++t) {
    // Issue this step's global loads early (latency hides under max/P/WMMA)
    float fv[8];
    int   mk[8];
#pragma unroll
    for (int v = 0; v < 8; ++v) {
      int M = v + 8 * khalf;
      size_t bt = (bbase + M) * TDIM + t;
      fv[v] = feats[bt * CDIM + ncol];
      mk[v] = maskp[bt];
      if (t + 1 < TDIM) __builtin_prefetch(feats + (bt + 1) * CDIM + ncol, 0, 0);
    }

    // Per-row max via in-wave shuffles; each thread owns 8 contiguous columns
    {
      v8f av = *(const v8f*)(&sAlpha[rrow * CDIM + rj * 8]);   // 2x ds_load_b128
      float mx = av[0];
#pragma unroll
      for (int c8 = 1; c8 < 8; ++c8) mx = fmaxf(mx, av[c8]);
#pragma unroll
      for (int m = 1; m <= 8; m <<= 1) mx = fmaxf(mx, __shfl_xor(mx, m, 32));
      if (rj == 0) sMrow[rrow] = mx;
      v8h ph;
#pragma unroll
      for (int c8 = 0; c8 < 8; ++c8)
        ph[c8] = (_Float16)__builtin_amdgcn_exp2f((av[c8] - mx) * LOG2E);
      *(v8h*)(&sP[rrow * EPAD + rj * 8]) = ph;                 // 1x ds_store_b128
    }
    __syncthreads();   // P, sMrow visible to all waves

    // Hoisted per-lane row maxima (one v8f = 2x ds_load_b128)
    v8f mrow8 = *(const v8f*)(&sMrow[8 * khalf]);

    // S(16x16 tile) = P(16x128) x E(128x128) via v_wmma_f32_16x16x32_f16
    v16h afr[4];
    {
      const _Float16* pr = &sP[nloc * EPAD];
#pragma unroll
      for (int kb = 0; kb < 4; ++kb) {
        v8h lo = *(const v8h*)(pr + kb * 32 + 8 * khalf);
        v8h hi = *(const v8h*)(pr + kb * 32 + 16 + 8 * khalf);
        afr[kb] = cat8(lo, hi);
      }
    }
    v8f acc = {0.f, 0.f, 0.f, 0.f, 0.f, 0.f, 0.f, 0.f};
#pragma unroll
    for (int kb = 0; kb < 4; ++kb)
      acc = __builtin_amdgcn_wmma_f32_16x16x32_f16(
          false, afr[kb], false, bfrag[kb], (short)0, acc, false, false);

    // alpha_new = feat + m + ln2*log2(S); bit-select keeps old where mask==0
#pragma unroll
    for (int v = 0; v < 8; ++v) {
      int M = v + 8 * khalf;                      // C/D layout: VGPR v, khalf -> row
      float nv = fv[v] + mrow8[v] + LN2 * __builtin_amdgcn_logf(acc[v]);
      alphaReg[v] = bitsel(mk[v], nv, alphaReg[v]);
      sAlpha[M * CDIM + ncol] = alphaReg[v];      // unconditional mirror store
    }
    __syncthreads();   // alpha visible for next step's reductions
  }

  // Final: logsumexp_i(alpha_i + trans[i, STOP]) per row via shuffles
  {
    v8f av = *(const v8f*)(&sAlpha[rrow * CDIM + rj * 8]);
    float tv[8];
#pragma unroll
    for (int c8 = 0; c8 < 8; ++c8)
      tv[c8] = av[c8] + trans[(rj * 8 + c8) * CDIM + (CDIM - 1)];
    float mx = tv[0];
#pragma unroll
    for (int c8 = 1; c8 < 8; ++c8) mx = fmaxf(mx, tv[c8]);
#pragma unroll
    for (int m = 1; m <= 8; m <<= 1) mx = fmaxf(mx, __shfl_xor(mx, m, 32));
    float s = 0.0f;
#pragma unroll
    for (int c8 = 0; c8 < 8; ++c8)
      s += __builtin_amdgcn_exp2f((tv[c8] - mx) * LOG2E);
#pragma unroll
    for (int m = 1; m <= 8; m <<= 1) s += __shfl_xor(s, m, 32);
    if (rj == 0) atomicAdd(acc_out, mx + LN2 * __builtin_amdgcn_logf(s));
  }
}

// Gold path score: one block per batch row
__global__ __launch_bounds__(128) void crf_gold_score(
    const float* __restrict__ feats, const int* __restrict__ maskp,
    const int* __restrict__ tags, const float* __restrict__ trans,
    float* __restrict__ acc_out)
{
  __shared__ float red[128];
  const int b = blockIdx.x, tid = threadIdx.x;

  int lsum = 0;
  for (int t = tid; t < TDIM; t += 128) lsum += maskp[b * TDIM + t];
  red[tid] = (float)lsum;
  __syncthreads();
  for (int off = 64; off > 0; off >>= 1) {
    if (tid < off) red[tid] += red[tid + off];
    __syncthreads();
  }
  const int len = (int)red[0];   // valid length in [1, T]
  __syncthreads();

  float s = 0.0f;
  for (int t = tid; t < TDIM; t += 128) {
    if (t < len) {
      int tag = tags[b * TDIM + t];
      s += feats[((size_t)b * TDIM + t) * CDIM + tag];                 // emission
      int prev = (t == 0) ? (CDIM - 2) : tags[b * TDIM + t - 1];       // START or prev
      s += trans[prev * CDIM + tag];                                   // transition
    }
  }
  if (tid == 0)  // final transition tags[len-1] -> STOP
    s += trans[tags[b * TDIM + len - 1] * CDIM + (CDIM - 1)];
  red[tid] = s;
  __syncthreads();
  for (int off = 64; off > 0; off >>= 1) {
    if (tid < off) red[tid] += red[tid + off];
    __syncthreads();
  }
  if (tid == 0) atomicAdd(acc_out, red[0]);
}

__global__ void crf_init(float* __restrict__ ws) { ws[0] = 0.0f; ws[1] = 0.0f; }

__global__ void crf_finalize(const float* __restrict__ ws, float* __restrict__ out) {
  out[0] = (ws[0] - ws[1]) * (1.0f / (float)BDIM);
}

extern "C" void kernel_launch(void* const* d_in, const int* in_sizes, int n_in,
                              void* d_out, int out_size, void* d_ws, size_t ws_size,
                              hipStream_t stream) {
  (void)in_sizes; (void)n_in; (void)out_size; (void)ws_size;
  const float* feats = (const float*)d_in[0];
  const int*   maskp = (const int*)d_in[1];
  const int*   tags  = (const int*)d_in[2];
  const float* trans = (const float*)d_in[3];
  float* ws  = (float*)d_ws;
  float* out = (float*)d_out;

  crf_init<<<1, 1, 0, stream>>>(ws);
  crf_forward_scan<<<BDIM / ROWS, 256, 0, stream>>>(feats, maskp, trans, ws + 0);
  crf_gold_score<<<BDIM, 128, 0, stream>>>(feats, maskp, tags, trans, ws + 1);
  crf_finalize<<<1, 1, 0, stream>>>(ws, out);
}
